// MaskedSelfAttention_76940044140785
// MI455X (gfx1250) — matrix operand order repaired and checked by reading, not compiled, not run on hardware
//
#include <hip/hip_runtime.h>
#include <hip/hip_bf16.h>

// MI455X / gfx1250: wave32, WMMA 16x16x32 bf16 -> f32 accumulate.

typedef __attribute__((ext_vector_type(16))) __bf16 v16bf;
typedef __attribute__((ext_vector_type(8)))  float  v8f;
typedef __attribute__((ext_vector_type(4)))  float  f4;

#define BATCH 16
#define SEQ   1024
#define DIM   768
#define N3    2304   // 3*DIM
#define LDP   40     // padded LDS row: 80B (16B-aligned, conflict-free for 16 lanes)

__device__ __forceinline__ __bf16 f2bf(float f) {
    return (__bf16)f;      // RTNE, native convert on gfx1250
}

__device__ __forceinline__ v8f wmma_bf16(v16bf a, v16bf b, v8f c) {
    return __builtin_amdgcn_wmma_f32_16x16x32_bf16(
        /*neg_a=*/false, a, /*neg_b=*/false, b,
        /*c_mod=*/(short)0, c, /*reuse_a=*/false, /*reuse_b=*/false);
}

// A fragment: 16x32 bf16, row-major source with leading dim ld (ISA 7.12.2).
// lanes 0-15: M=lane, K = 0..7 (j<8) then 16..23 (j>=8)
// lanes 16-31: M=lane-16, K = 8..15 then 24..31
__device__ __forceinline__ v16bf load_a_frag(const __bf16* __restrict__ base, int ld, int lane) {
    int m  = lane & 15;
    int kb = (lane < 16) ? 0 : 8;
    const __bf16* p = base + (size_t)m * ld + kb;
    v16bf a;
#pragma unroll
    for (int j = 0; j < 8; ++j)  a[j] = p[j];
#pragma unroll
    for (int j = 8; j < 16; ++j) a[j] = p[8 + j];      // K = kb + 16 + (j-8)
    return a;
}

// B fragment 32x16 where B(kk,nn) = srcT[nn][kk] (source row-major [N][K]);
// per-lane data is two runs of 8 contiguous bf16 (K = kb2..kb2+7, kb2+16..kb2+23) -> two b128 loads.
__device__ __forceinline__ v16bf load_b_transposed(const __bf16* __restrict__ srcT, int ld,
                                                   int n0, int k0, int lane) {
    int n   = n0 + (lane & 15);
    int kb2 = (lane < 16) ? 0 : 8;
    const __bf16* p = srcT + (size_t)n * ld + k0 + kb2;
    v16bf b;
#pragma unroll
    for (int j = 0; j < 8; ++j)  b[j] = p[j];
#pragma unroll
    for (int j = 8; j < 16; ++j) b[j] = p[8 + j];
    return b;
}

// ---------------------------------------------------------------------------
// Kernel 1: qkv = x @ W, fp32 in -> bf16 out. q,k row-major; v stored
// TRANSPOSED per batch (vT[dim][seq]) so the PV GEMM gets contiguous B-frags.
// 256 thr (8 waves), tile 128(M) x 128(N), K-chunks of 32, register
// double-buffered global->LDS staging.
// ---------------------------------------------------------------------------
__global__ __launch_bounds__(256) void qkv_gemm(const float* __restrict__ x,
                                                const float* __restrict__ W,
                                                __bf16* __restrict__ qb,
                                                __bf16* __restrict__ kb,
                                                __bf16* __restrict__ vbT) {
    __shared__ __bf16 As [128][LDP];   // x tile, row-major [m][k]
    __shared__ __bf16 BsT[128][LDP];   // W tile, TRANSPOSED [n][k]

    const int tid  = threadIdx.x;
    const int lane = tid & 31;
    const int wave = tid >> 5;
    const int m0   = blockIdx.y * 128;
    const int n1   = blockIdx.x * 128;
    const int wr   = wave >> 1;   // 0..3 -> 32 rows each
    const int wc   = wave & 1;    // 0..1 -> 64 cols each

    v8f c[2][4];
#pragma unroll
    for (int i = 0; i < 2; ++i)
#pragma unroll
        for (int j = 0; j < 4; ++j) c[i][j] = {};

    // register-staged double buffer: preload K-chunk 0
    f4 av[4], bv[4];
#pragma unroll
    for (int i = 0; i < 4; ++i) {
        int idx4 = i * 256 + tid;
        av[i] = *(const f4*)&x[(size_t)(m0 + (idx4 >> 3)) * DIM + ((idx4 & 7) * 4)];
        bv[i] = *(const f4*)&W[(size_t)(idx4 >> 5) * N3 + n1 + ((idx4 & 31) * 4)];
    }

    for (int kt = 0; kt < DIM; kt += 32) {
        // spill staged registers into LDS (convert to bf16; B transposed)
#pragma unroll
        for (int i = 0; i < 4; ++i) {
            int idx4 = i * 256 + tid;
            int ra = idx4 >> 3, ca = (idx4 & 7) * 4;
            int rb = idx4 >> 5, cb = (idx4 & 31) * 4;
#pragma unroll
            for (int e = 0; e < 4; ++e) {
                As[ra][ca + e]  = f2bf(av[i][e]);
                BsT[cb + e][rb] = f2bf(bv[i][e]);
            }
        }
        __syncthreads();

        // issue next chunk's global loads before touching LDS
        if (kt + 32 < DIM) {
#pragma unroll
            for (int i = 0; i < 4; ++i) {
                int idx4 = i * 256 + tid;
                av[i] = *(const f4*)&x[(size_t)(m0 + (idx4 >> 3)) * DIM + kt + 32 + ((idx4 & 7) * 4)];
                bv[i] = *(const f4*)&W[(size_t)(kt + 32 + (idx4 >> 5)) * N3 + n1 + ((idx4 & 31) * 4)];
            }
        }

        v16bf af[2];
        af[0] = load_a_frag(&As[wr * 32][0],      LDP, lane);
        af[1] = load_a_frag(&As[wr * 32 + 16][0], LDP, lane);
#pragma unroll
        for (int nf = 0; nf < 4; ++nf) {
            v16bf bf = load_b_transposed(&BsT[0][0], LDP, wc * 64 + nf * 16, 0, lane);
            c[0][nf] = wmma_bf16(af[0], bf, c[0][nf]);
            c[1][nf] = wmma_bf16(af[1], bf, c[1][nf]);
        }
        __syncthreads();
    }

    // epilogue: C layout M = r + 8*(lane>=16), N = lane&15
    const int mb = (lane < 16) ? 0 : 8;
#pragma unroll
    for (int mf = 0; mf < 2; ++mf)
#pragma unroll
        for (int nf = 0; nf < 4; ++nf)
#pragma unroll
            for (int r = 0; r < 8; ++r) {
                int row = m0 + wr * 32 + mf * 16 + mb + r;   // = b*1024 + s
                int col = n1 + wc * 64 + nf * 16 + (lane & 15);
                __bf16 h = f2bf(c[mf][nf][r]);
                if (col < DIM) {
                    qb[(size_t)row * DIM + col] = h;
                } else if (col < 2 * DIM) {
                    kb[(size_t)row * DIM + (col - DIM)] = h;
                } else {
                    int d = col - 2 * DIM, bidx = row >> 10, s = row & 1023;
                    vbT[((size_t)bidx * DIM + d) * SEQ + s] = h;   // transposed
                }
            }
}

// ---------------------------------------------------------------------------
// Kernel 2: scores = mask(scale * q @ k^T) -> d_out; per-row (max, sumexp) -> ws
// Workgroup: (batch, 64 q-rows). 8 waves = 4 q-tiles x 2 key-halves.
// ---------------------------------------------------------------------------
__global__ __launch_bounds__(256) void attn_scores(const __bf16* __restrict__ qb,
                                                   const __bf16* __restrict__ kb,
                                                   float* __restrict__ scores,
                                                   float* __restrict__ smax,
                                                   float* __restrict__ ssum) {
    const int q0 = blockIdx.x * 64;
    const int b  = blockIdx.y;
    const __bf16* qm = qb + (size_t)b * SEQ * DIM;
    const __bf16* km = kb + (size_t)b * SEQ * DIM;
    float* sc = scores + (size_t)b * SEQ * SEQ;

    const int tid   = threadIdx.x;
    const int lane  = tid & 31;
    const int wave  = tid >> 5;
    const int qtile = wave >> 1;
    const int kh    = wave & 1;
    const int mb    = (lane < 16) ? 0 : 8;
    const int ncol  = lane & 15;
    const int qr0   = q0 + qtile * 16;
    const float scale = 0.03608439182435161f;   // 1/sqrt(768)
    const float NEG   = -__builtin_inff();

    float lm[8], ls[8];
#pragma unroll
    for (int r = 0; r < 8; ++r) { lm[r] = NEG; ls[r] = 0.0f; }

    const int nIter = q0 / 32 + 2;              // key tiles while kstart <= q0+63
    for (int it = 0; it < nIter; ++it) {
        const int kstart = it * 32 + kh * 16;
        if (kstart > qr0 + 15) {                // tile entirely above diagonal
#pragma unroll
            for (int r = 0; r < 8; ++r)
                sc[(size_t)(qr0 + mb + r) * SEQ + kstart + ncol] = NEG;
            continue;
        }
        v8f c = {};
        for (int d0 = 0; d0 < DIM; d0 += 32) {
            v16bf a  = load_a_frag(qm + (size_t)qr0 * DIM + d0, DIM, lane);
            v16bf bf = load_b_transposed(km, DIM, kstart, d0, lane);
            c = wmma_bf16(a, bf, c);
        }
#pragma unroll
        for (int r = 0; r < 8; ++r) {
            int qi = qr0 + mb + r;
            int kj = kstart + ncol;
            bool ok = (kj <= qi);
            float val = ok ? c[r] * scale : NEG;
            sc[(size_t)qi * SEQ + kj] = val;
            if (ok) {                            // guarded online max/sum
                float nm = fmaxf(lm[r], val);
                ls[r] = ls[r] * __expf(lm[r] - nm) + __expf(val - nm);
                lm[r] = nm;
            }
        }
    }

    // cross-lane (16 lanes share a row-set) flash combine, -inf-safe
#pragma unroll
    for (int off = 1; off < 16; off <<= 1) {
#pragma unroll
        for (int r = 0; r < 8; ++r) {
            float m2 = __shfl_xor(lm[r], off);
            float s2 = __shfl_xor(ls[r], off);
            float nm = fmaxf(lm[r], m2);
            float e1 = (lm[r] == NEG) ? 0.0f : __expf(lm[r] - nm);
            float e2 = (m2    == NEG) ? 0.0f : __expf(m2    - nm);
            ls[r] = ls[r] * e1 + s2 * e2;
            lm[r] = nm;
        }
    }

    __shared__ float redM[64][2];
    __shared__ float redS[64][2];
    if ((lane & 15) == 0) {
#pragma unroll
        for (int r = 0; r < 8; ++r) {
            redM[qtile * 16 + mb + r][kh] = lm[r];
            redS[qtile * 16 + mb + r][kh] = ls[r];
        }
    }
    __syncthreads();

    if (tid < 64) {                              // combine the two key-halves
        float m0v = redM[tid][0], m1v = redM[tid][1];
        float nm = fmaxf(m0v, m1v);
        float e0 = (m0v == NEG) ? 0.0f : __expf(m0v - nm);
        float e1 = (m1v == NEG) ? 0.0f : __expf(m1v - nm);
        smax[b * SEQ + q0 + tid] = nm;
        ssum[b * SEQ + q0 + tid] = redS[tid][0] * e0 + redS[tid][1] * e1;
    }

    // fill fully-masked region: columns >= q0+64
    const int colstart = q0 + 64;
    const int ncols = SEQ - colstart;
    for (int idx = tid; idx < 64 * ncols; idx += 256) {
        int rr = idx / ncols;
        int cc = colstart + (idx - rr * ncols);
        sc[(size_t)(q0 + rr) * SEQ + cc] = NEG;
    }
}

// ---------------------------------------------------------------------------
// Kernel 3: context = softmax(scores) @ v, probabilities rebuilt on the fly.
// v is TRANSPOSED (vT[dim][seq]) -> contiguous b128 B-fragment loads.
// Workgroup: 8 waves, one 16-row q-tile x 128 output dims; triangular K-loop.
// ---------------------------------------------------------------------------
__global__ __launch_bounds__(256) void attn_context(const float* __restrict__ scores,
                                                    const __bf16* __restrict__ vbT,
                                                    const float* __restrict__ smax,
                                                    const float* __restrict__ ssum,
                                                    float* __restrict__ outc) {
    const int b    = blockIdx.z;
    const int qt   = blockIdx.y * 16;
    const int wave = threadIdx.x >> 5;
    const int lane = threadIdx.x & 31;
    const int n0   = blockIdx.x * 128 + wave * 16;

    const float*  sc  = scores + (size_t)b * SEQ * SEQ;
    const __bf16* vmT = vbT    + (size_t)b * DIM * SEQ;

    const int   rowA = qt + (lane & 15);          // A-fragment row for this lane
    const float mx   = smax[b * SEQ + rowA];
    const float rs   = 1.0f / ssum[b * SEQ + rowA];
    const int   kb2  = (lane < 16) ? 0 : 8;

    v8f c = {};
    for (int k0 = 0; k0 < qt + 16; k0 += 32) {    // keys beyond the tile give p=0
        const float* sp = sc + (size_t)rowA * SEQ + k0 + kb2;
        v16bf a;                                   // P = exp(s-mx)/sum ; exp(-inf)=0
#pragma unroll
        for (int j = 0; j < 8; ++j)  a[j] = f2bf(__expf(sp[j]     - mx) * rs);
#pragma unroll
        for (int j = 8; j < 16; ++j) a[j] = f2bf(__expf(sp[8 + j] - mx) * rs);
        v16bf bf = load_b_transposed(vmT, SEQ, n0, k0, lane);   // B(kk,nn)=vT[nn][kk]
        c = wmma_bf16(a, bf, c);
    }

    const int colN = n0 + (lane & 15);
    const int mb   = (lane < 16) ? 0 : 8;
#pragma unroll
    for (int r = 0; r < 8; ++r) {
        int qi = qt + mb + r;
        outc[((size_t)b * SEQ + qi) * DIM + colN] = c[r];
    }
}

// ---------------------------------------------------------------------------
extern "C" void kernel_launch(void* const* d_in, const int* in_sizes, int n_in,
                              void* d_out, int out_size, void* d_ws, size_t ws_size,
                              hipStream_t stream) {
    const float* x = (const float*)d_in[0];   // [16,1024,768]
    const float* W = (const float*)d_in[1];   // [768,2304]

    float* out    = (float*)d_out;
    float* ctx    = out;                                      // 16*1024*768
    float* scores = out + (size_t)BATCH * SEQ * DIM;          // 16*1024*1024

    __bf16* qb   = (__bf16*)d_ws;
    __bf16* kb   = qb + (size_t)BATCH * SEQ * DIM;
    __bf16* vbT  = kb + (size_t)BATCH * SEQ * DIM;            // [b][dim][seq]
    float*  smax = (float*)(vbT + (size_t)BATCH * SEQ * DIM);
    float*  ssum = smax + BATCH * SEQ;

    qkv_gemm     <<<dim3(N3 / 128, (BATCH * SEQ) / 128), 256, 0, stream>>>(x, W, qb, kb, vbT);
    attn_scores  <<<dim3(SEQ / 64, BATCH),               256, 0, stream>>>(qb, kb, scores, smax, ssum);
    attn_context <<<dim3(DIM / 128, SEQ / 16, BATCH),    256, 0, stream>>>(scores, vbT, smax, ssum, ctx);
}
